// Temporal_Attention_36971078484297
// MI455X (gfx1250) — matrix operand order repaired and checked by reading, not compiled, not run on hardware
//
#include <hip/hip_runtime.h>
#include <hip/hip_bf16.h>
#include <stdint.h>

// ---- problem dims (fixed by the reference) ----
#define B_DIM 16
#define C_DIM 512
#define N_DIM 4096
#define BN_EPS 1e-5f

typedef __bf16 bf16_t;
typedef __bf16 v16bf __attribute__((ext_vector_type(16)));
typedef float  v8f   __attribute__((ext_vector_type(8)));

union FragU { uint4 q[2]; v16bf v; };

__device__ __forceinline__ unsigned short f2bf(float f) {
    unsigned int u = __float_as_uint(f);
    u += 0x7FFFu + ((u >> 16) & 1u);      // round-to-nearest-even
    return (unsigned short)(u >> 16);
}

// ---------------------------------------------------------------------------
// x [B,C,N] f32  ->  xT [B,N,C] bf16   (LDS-tiled transpose + convert)
// ---------------------------------------------------------------------------
__global__ __launch_bounds__(256) void transpose_to_bf16(
        const float* __restrict__ x, unsigned short* __restrict__ xT) {
    __shared__ unsigned short tile[32][33];
    int b  = blockIdx.z;
    int n0 = blockIdx.x * 32;
    int c0 = blockIdx.y * 32;
    int tx = threadIdx.x;   // 0..31
    int ty = threadIdx.y;   // 0..7
    const float* xb = x + (size_t)b * C_DIM * N_DIM;
    unsigned short* xTb = xT + (size_t)b * N_DIM * C_DIM;
    #pragma unroll
    for (int j = 0; j < 32; j += 8)
        tile[ty + j][tx] = f2bf(xb[(size_t)(c0 + ty + j) * N_DIM + n0 + tx]);
    __syncthreads();
    #pragma unroll
    for (int j = 0; j < 32; j += 8)
        xTb[(size_t)(n0 + ty + j) * C_DIM + c0 + tx] = tile[tx][ty + j];
}

// ---------------------------------------------------------------------------
// Wq/Wk/Wv f32 [C,C] -> bf16
// ---------------------------------------------------------------------------
__global__ void convert_w(const float* __restrict__ Wq, const float* __restrict__ Wk,
                          const float* __restrict__ Wv,
                          unsigned short* __restrict__ Wqb, unsigned short* __restrict__ Wkb,
                          unsigned short* __restrict__ Wvb, int n) {
    int i = blockIdx.x * blockDim.x + threadIdx.x;
    if (i < n) { Wqb[i] = f2bf(Wq[i]); Wkb[i] = f2bf(Wk[i]); Wvb[i] = f2bf(Wv[i]); }
}

// ---------------------------------------------------------------------------
// Fragment loads straight from global per the CDNA5 WMMA VGPR layouts.
// A row-major [M,K]: lane l holds row m; K chunks {k0+hi*8..+7} and {k0+16+hi*8..+7}.
// BT row-major [N,K]: lane l holds column n; K chunks {k0+hi*8..+7} and {k0+16+hi*8..+7}.
// ---------------------------------------------------------------------------
__device__ __forceinline__ void load_ab(
        const unsigned short* __restrict__ ap0, const unsigned short* __restrict__ ap1,
        const unsigned short* const bp[4], int k0, FragU a[2], FragU b[4]) {
    a[0].q[0] = *(const uint4*)(ap0 + k0);
    a[0].q[1] = *(const uint4*)(ap0 + k0 + 16);
    a[1].q[0] = *(const uint4*)(ap1 + k0);
    a[1].q[1] = *(const uint4*)(ap1 + k0 + 16);
    #pragma unroll
    for (int t = 0; t < 4; ++t) {
        b[t].q[0] = *(const uint4*)(bp[t] + k0);
        b[t].q[1] = *(const uint4*)(bp[t] + k0 + 16);
    }
}

__device__ __forceinline__ void mma_tile(const FragU a[2], const FragU b[4], v8f acc[2][4]) {
    #pragma unroll
    for (int i = 0; i < 2; ++i)
        #pragma unroll
        for (int t = 0; t < 4; ++t)
            acc[i][t] = __builtin_amdgcn_wmma_f32_16x16x32_bf16(
                false, a[i].v, false, b[t].v, (short)0, acc[i][t], false, false);
}

// ---------------------------------------------------------------------------
// Generic WMMA GEMM:  D[M,N] = A[M,K] (row-major bf16) x BT[N,K]^T (row-major bf16)
// Leading dims of A and BT are both K (true for every call site here).
// mode 0: store bf16 row-major (ldd)       mode 1: store bf16 transposed (ldd = lead of D^T)
// mode 2: store f32 row-major (+ optional residual add, f32)
// Block: 256 threads = 8 waves (4 in M x 2 in N); wave tile 32x64; block tile 128x128.
// K must be a multiple of 64 (512 and 4096 here). Register double-buffered K pipeline.
// ---------------------------------------------------------------------------
__global__ __launch_bounds__(256) void wmma_gemm_bf16(
        const unsigned short* __restrict__ A,  long long strideA,
        const unsigned short* __restrict__ BT, long long strideB,
        void* __restrict__ D, long long strideD, int ldd,
        const float* __restrict__ resid, long long strideR, int ldr,
        int K, int mode) {
    int b = blockIdx.z;
    const unsigned short* Ab = A  + (size_t)b * strideA;
    const unsigned short* Bb = BT + (size_t)b * strideB;

    int wave = threadIdx.x >> 5;
    int lane = threadIdx.x & 31;
    int l  = lane & 15;
    int hi = lane >> 4;
    int wm = wave & 3;          // 4 waves along M
    int wn = wave >> 2;         // 2 waves along N

    int m0 = blockIdx.x * 128 + wm * 32;
    int n0 = blockIdx.y * 128 + wn * 64;

    v8f acc[2][4];
    #pragma unroll
    for (int i = 0; i < 2; ++i)
        #pragma unroll
        for (int t = 0; t < 4; ++t)
            #pragma unroll
            for (int g = 0; g < 8; ++g) acc[i][t][g] = 0.0f;

    const unsigned short* ap0 = Ab + (size_t)(m0 + l) * K + hi * 8;
    const unsigned short* ap1 = Ab + (size_t)(m0 + 16 + l) * K + hi * 8;
    const unsigned short* bp[4];
    #pragma unroll
    for (int t = 0; t < 4; ++t)
        bp[t] = Bb + (size_t)(n0 + t * 16 + l) * K + hi * 8;

    FragU fa0[2], fb0[4], fa1[2], fb1[4];
    load_ab(ap0, ap1, bp, 0, fa0, fb0);
    for (int k0 = 0; k0 < K; k0 += 64) {
        load_ab(ap0, ap1, bp, k0 + 32, fa1, fb1);     // prefetch next half-step
        mma_tile(fa0, fb0, acc);
        if (k0 + 64 < K)
            load_ab(ap0, ap1, bp, k0 + 64, fa0, fb0); // prefetch step after that
        mma_tile(fa1, fb1, acc);
    }

    // C/D layout: VGPR g -> row m0 + i*16 + g + hi*8, column n0 + t*16 + l
    #pragma unroll
    for (int i = 0; i < 2; ++i) {
        #pragma unroll
        for (int t = 0; t < 4; ++t) {
            int n = n0 + t * 16 + l;
            #pragma unroll
            for (int g = 0; g < 8; ++g) {
                int m = m0 + i * 16 + g + hi * 8;
                float val = acc[i][t][g];
                if (mode == 0) {
                    ((unsigned short*)D)[(size_t)b * strideD + (size_t)m * ldd + n] = f2bf(val);
                } else if (mode == 1) {
                    ((unsigned short*)D)[(size_t)b * strideD + (size_t)n * ldd + m] = f2bf(val);
                } else {
                    float r = resid ? resid[(size_t)b * strideR + (size_t)m * ldr + n] : 0.0f;
                    ((float*)D)[(size_t)b * strideD + (size_t)m * ldd + n] = val + r;
                }
            }
        }
    }
}

// ---------------------------------------------------------------------------
// Row softmax over last dim (C=512): S f32 [B*C, C] -> P bf16 [B*C, C]
// ---------------------------------------------------------------------------
__global__ __launch_bounds__(256) void softmax_rows(
        const float* __restrict__ S, unsigned short* __restrict__ P) {
    int row = blockIdx.x;
    const float* s = S + (size_t)row * C_DIM;
    unsigned short* p = P + (size_t)row * C_DIM;
    int tid = threadIdx.x;
    float a = s[tid], b = s[tid + 256];
    __shared__ float red[256];
    red[tid] = fmaxf(a, b);
    __syncthreads();
    for (int off = 128; off > 0; off >>= 1) {
        if (tid < off) red[tid] = fmaxf(red[tid], red[tid + off]);
        __syncthreads();
    }
    float mx = red[0];
    __syncthreads();
    float e0 = __expf(a - mx), e1 = __expf(b - mx);
    red[tid] = e0 + e1;
    __syncthreads();
    for (int off = 128; off > 0; off >>= 1) {
        if (tid < off) red[tid] += red[tid + off];
        __syncthreads();
    }
    float inv = 1.0f / red[0];
    p[tid] = f2bf(e0 * inv);
    p[tid + 256] = f2bf(e1 * inv);
}

// ---------------------------------------------------------------------------
// BatchNorm batch stats: y [B,C,N] -> stats[c]=mean, stats[C+c]=var
// (deterministic tree reduce; no float atomics so replays are bit-identical)
// ---------------------------------------------------------------------------
__global__ __launch_bounds__(256) void bn_stats(
        const float* __restrict__ y, float* __restrict__ stats) {
    int c = blockIdx.x, tid = threadIdx.x;
    float s = 0.0f, s2 = 0.0f;
    for (int b = 0; b < B_DIM; ++b) {
        const float* yb = y + ((size_t)b * C_DIM + c) * N_DIM;
        for (int n = tid; n < N_DIM; n += 256) {
            float v = yb[n];
            s += v; s2 += v * v;
        }
    }
    __shared__ float rs[256], rs2[256];
    rs[tid] = s; rs2[tid] = s2;
    __syncthreads();
    for (int off = 128; off > 0; off >>= 1) {
        if (tid < off) { rs[tid] += rs[tid + off]; rs2[tid] += rs2[tid + off]; }
        __syncthreads();
    }
    if (tid == 0) {
        float inv = 1.0f / ((float)B_DIM * (float)N_DIM);
        float mean = rs[0] * inv;
        stats[c] = mean;
        stats[C_DIM + c] = rs2[0] * inv - mean * mean;
    }
}

// ---------------------------------------------------------------------------
// BN apply: out = (y - mean) * rsqrt(var+eps) * gamma + beta
// ---------------------------------------------------------------------------
__global__ void bn_apply(const float* __restrict__ y, const float* __restrict__ stats,
                         const float* __restrict__ gamma, const float* __restrict__ beta,
                         float* __restrict__ out) {
    size_t i = (size_t)blockIdx.x * blockDim.x + threadIdx.x;
    size_t total = (size_t)B_DIM * C_DIM * N_DIM;
    if (i >= total) return;
    int c = (int)((i / N_DIM) % C_DIM);
    float sc = rsqrtf(stats[C_DIM + c] + BN_EPS) * gamma[c];
    out[i] = (y[i] - stats[c]) * sc + beta[c];
}

// ---------------------------------------------------------------------------
extern "C" void kernel_launch(void* const* d_in, const int* in_sizes, int n_in,
                              void* d_out, int out_size, void* d_ws, size_t ws_size,
                              hipStream_t stream) {
    typedef long long ll;
    const float* x     = (const float*)d_in[0];
    const float* Wq    = (const float*)d_in[1];
    const float* Wk    = (const float*)d_in[2];
    const float* Wv    = (const float*)d_in[3];
    const float* gamma = (const float*)d_in[4];
    const float* beta  = (const float*)d_in[5];
    float* out = (float*)d_out;

    char* ws = (char*)d_ws;
    size_t off = 0;
    auto alloc = [&](size_t bytes) -> char* {
        char* p = ws + off;
        off += (bytes + 255) & ~(size_t)255;
        return p;
    };
    const size_t BCN = (size_t)B_DIM * C_DIM * N_DIM;
    const size_t BCC = (size_t)B_DIM * C_DIM * C_DIM;
    unsigned short* xT    = (unsigned short*)alloc(BCN * 2);                 // [B,N,C] bf16
    unsigned short* Wqb   = (unsigned short*)alloc((size_t)C_DIM * C_DIM * 2);
    unsigned short* Wkb   = (unsigned short*)alloc((size_t)C_DIM * C_DIM * 2);
    unsigned short* Wvb   = (unsigned short*)alloc((size_t)C_DIM * C_DIM * 2);
    unsigned short* Q     = (unsigned short*)alloc(BCN * 2);                 // [B,C,N] bf16
    unsigned short* Km    = (unsigned short*)alloc(BCN * 2);                 // [B,C,N] bf16
    unsigned short* VT    = (unsigned short*)alloc(BCN * 2);                 // [B,N,C] bf16
    float*          S     = (float*)alloc(BCC * 4);                          // [B,C,C] f32
    unsigned short* P     = (unsigned short*)alloc(BCC * 2);                 // [B,C,C] bf16
    float*          Y     = (float*)alloc(BCN * 4);                          // [B,C,N] f32
    float*          stats = (float*)alloc((size_t)C_DIM * 2 * 4);

    // 1) prep: transpose/convert x, convert weights
    transpose_to_bf16<<<dim3(N_DIM / 32, C_DIM / 32, B_DIM), dim3(32, 8), 0, stream>>>(x, xT);
    convert_w<<<(C_DIM * C_DIM + 255) / 256, 256, 0, stream>>>(
        Wq, Wk, Wv, Wqb, Wkb, Wvb, C_DIM * C_DIM);

    dim3 blk(256);
    dim3 gCN(C_DIM / 128, N_DIM / 128, B_DIM);  // M=512, N=4096 tiles (128x128 blocks)
    dim3 gCC(C_DIM / 128, C_DIM / 128, B_DIM);  // M=512, N=512  tiles

    // 2) Q = Wq * x   (A = Wq [C,C], BT = xT [N,C])  -> bf16 [C,N]
    wmma_gemm_bf16<<<gCN, blk, 0, stream>>>(Wqb, 0, xT, (ll)N_DIM * C_DIM,
        Q, (ll)C_DIM * N_DIM, N_DIM, nullptr, 0, 0, C_DIM, 0);
    // 3) K = Wk * x -> bf16 [C,N]
    wmma_gemm_bf16<<<gCN, blk, 0, stream>>>(Wkb, 0, xT, (ll)N_DIM * C_DIM,
        Km, (ll)C_DIM * N_DIM, N_DIM, nullptr, 0, 0, C_DIM, 0);
    // 4) V = Wv * x -> stored transposed bf16 [N,C]
    wmma_gemm_bf16<<<gCN, blk, 0, stream>>>(Wvb, 0, xT, (ll)N_DIM * C_DIM,
        VT, (ll)N_DIM * C_DIM, C_DIM, nullptr, 0, 0, C_DIM, 1);
    // 5) S = Q * K^T  (A = Q [C,N], BT = K [C,N], Kdim = N) -> f32 [C,C]
    wmma_gemm_bf16<<<gCC, blk, 0, stream>>>(Q, (ll)C_DIM * N_DIM, Km, (ll)C_DIM * N_DIM,
        S, (ll)C_DIM * C_DIM, C_DIM, nullptr, 0, 0, N_DIM, 2);
    // 6) P = softmax(S) -> bf16 [C,C]
    softmax_rows<<<B_DIM * C_DIM, 256, 0, stream>>>(S, P);
    // 7) Y = P * V + x  (A = P [C,C], BT = VT [N,C], residual x) -> f32 [C,N]
    wmma_gemm_bf16<<<gCN, blk, 0, stream>>>(P, (ll)C_DIM * C_DIM, VT, (ll)N_DIM * C_DIM,
        Y, (ll)C_DIM * N_DIM, N_DIM, x, (ll)C_DIM * N_DIM, N_DIM, C_DIM, 2);
    // 8) BatchNorm training-mode stats + apply
    bn_stats<<<C_DIM, 256, 0, stream>>>(Y, stats);
    bn_apply<<<(int)((BCN + 255) / 256), 256, 0, stream>>>(Y, stats, gamma, beta, out);
}
